// NodeNet_738734375749
// MI455X (gfx1250) — compile-verified, matrix operand order checked by reading, not run on hardware
//
#include <hip/hip_runtime.h>

typedef __attribute__((ext_vector_type(16))) _Float16 v16h;
typedef __attribute__((ext_vector_type(8)))  float    v8f;
typedef _Float16 half_t;

#define G_     4096
#define ODE_   64
#define NDATA_ 32
#define H_     256
#define EA_    64
#define EPG_   128
#define E_     (G_*EPG_)

// ---------------------------------------------------------------------------
// Weight prep: W (K x N, f32, row-major) -> Wt (N x K, fp16, row-major)
// so B-matrix fragments load as contiguous (K,K+1) fp16 pairs per lane.
// ---------------------------------------------------------------------------
__global__ void prep_wt(const float* __restrict__ W, half_t* __restrict__ Wt,
                        int K, int N) {
    int t = blockIdx.x * blockDim.x + threadIdx.x;
    if (t < K * N) {
        int n = t / K, k = t - n * K;
        Wt[t] = (half_t)W[k * N + n];
    }
}

// ---------------------------------------------------------------------------
// Fragment loader: 16-bit 16x32 A/B layout (CDNA5 ISA 7.12.2).
// lane L: row = L&15, halfwave = L>>4; VGPR v holds K pair:
//   K = ((v&4)<<2) + (halfwave<<3) + ((v&3)<<1)  (+pos)
// base is row-major with leading dim `ld` (halves); ld and k0 even -> 4B loads.
// ---------------------------------------------------------------------------
__device__ __forceinline__ v16h load_frag(const half_t* __restrict__ base,
                                          int ld, int row0, int k0, int lane) {
    int hl = lane >> 4, mr = lane & 15;
    const half_t* p = base + (row0 + mr) * ld + k0;
    union { v16h h; unsigned u[8]; } f;
#pragma unroll
    for (int v = 0; v < 8; ++v) {
        int K = ((v & 4) << 2) + (hl << 3) + ((v & 3) << 1);
        f.u[v] = *(const unsigned*)(p + K);
    }
    return f.h;
}

// Two row-tiles (m0a, m0b) x one col-tile (n0), shared B fragment.
__device__ __forceinline__ void gemm_pair(const half_t* __restrict__ A, int ldA,
                                          const half_t* __restrict__ Wt, int ldK,
                                          int m0a, int m0b, int n0, int Kdim,
                                          int lane, v8f& ca, v8f& cb) {
    for (int k0 = 0; k0 < Kdim; k0 += 32) {
        v16h b  = load_frag(Wt, ldK, n0, k0, lane);
        v16h aa = load_frag(A, ldA, m0a, k0, lane);
        v16h ab = load_frag(A, ldA, m0b, k0, lane);
        ca = __builtin_amdgcn_wmma_f32_16x16x32_f16(false, aa, false, b,
                                                    (short)0, ca, false, false);
        cb = __builtin_amdgcn_wmma_f32_16x16x32_f16(false, ab, false, b,
                                                    (short)0, cb, false, false);
    }
}

// Four row-tiles (rows 0,16,32,48) x one col-tile (n0): one B fragment feeds
// four WMMAs per k-step -> halves global weight traffic, 4 indep accumulators.
__device__ __forceinline__ void gemm_quad(const half_t* __restrict__ A, int ldA,
                                          const half_t* __restrict__ Wt, int ldK,
                                          int n0, int Kdim, int lane,
                                          v8f& c0, v8f& c1, v8f& c2, v8f& c3) {
    for (int k0 = 0; k0 < Kdim; k0 += 32) {
        v16h b  = load_frag(Wt, ldK, n0, k0, lane);
        v16h a0 = load_frag(A, ldA, 0,  k0, lane);
        v16h a1 = load_frag(A, ldA, 16, k0, lane);
        v16h a2 = load_frag(A, ldA, 32, k0, lane);
        v16h a3 = load_frag(A, ldA, 48, k0, lane);
        c0 = __builtin_amdgcn_wmma_f32_16x16x32_f16(false, a0, false, b,
                                                    (short)0, c0, false, false);
        c1 = __builtin_amdgcn_wmma_f32_16x16x32_f16(false, a1, false, b,
                                                    (short)0, c1, false, false);
        c2 = __builtin_amdgcn_wmma_f32_16x16x32_f16(false, a2, false, b,
                                                    (short)0, c2, false, false);
        c3 = __builtin_amdgcn_wmma_f32_16x16x32_f16(false, a3, false, b,
                                                    (short)0, c3, false, false);
    }
}

// C/D layout: lane holds col n0+(lane&15); VGPR r holds row m0 + r + 8*(lane>>4).
__device__ __forceinline__ void store_tile(half_t* __restrict__ Hs, int ldn,
                                           int m0, int n0, int lane,
                                           const v8f& c,
                                           const float* __restrict__ bias,
                                           bool relu) {
    int hl = lane >> 4, mr = lane & 15;
    int n = n0 + mr;
    float bv = bias[n];
#pragma unroll
    for (int r = 0; r < 8; ++r) {
        float v = c[r] + bv;
        if (relu) v = fmaxf(v, 0.f);
        Hs[(m0 + r + hl * 8) * ldn + n] = (half_t)v;
    }
}

// ---------------------------------------------------------------------------
// Phase 1: per-graph node MLP + row reduction -> feH (G x 64, fp16)
// ---------------------------------------------------------------------------
__global__ __launch_bounds__(256)
void node_mlp(const float* __restrict__ x,
              const half_t* __restrict__ Wt1, const half_t* __restrict__ Wt2,
              const half_t* __restrict__ Wt3,
              const float* __restrict__ b1, const float* __restrict__ b2,
              const float* __restrict__ b3,
              half_t* __restrict__ feH) {
    extern __shared__ char smem[];
    half_t* Ah = (half_t*)smem;        // 32 x 128
    half_t* H1 = Ah + 32 * 128;        // 32 x 256
    half_t* H2 = H1 + 32 * 256;        // 32 x 256

    int g = blockIdx.x;
    int tid = threadIdx.x, lane = tid & 31, wave = tid >> 5;

    // dp[n][2o+s] = x[(g*64+o)*64 + s*32 + n]; iterate in x memory order.
    const float* xg = x + (size_t)g * (ODE_ * 2 * NDATA_);
    for (int idx = tid; idx < ODE_ * 2 * NDATA_; idx += 256) {
        int o = idx >> 6, j = idx & 63;
        float v = xg[idx];
        int n = j & 31, c = (o << 1) | (j >> 5);
        Ah[n * 128 + c] = (half_t)v;
    }
    __syncthreads();

    // GEMM1: (32x128)x(128x256) +b1, relu -> H1.  16 col tiles / 8 waves.
    for (int j = 0; j < 2; ++j) {
        int n0 = (wave * 2 + j) * 16;
        v8f ca = {}, cb = {};
        gemm_pair(Ah, 128, Wt1, 128, 0, 16, n0, 128, lane, ca, cb);
        store_tile(H1, 256, 0, n0, lane, ca, b1, true);
        store_tile(H1, 256, 16, n0, lane, cb, b1, true);
    }
    __syncthreads();

    // GEMM2: (32x256)x(256x256) +b2, relu -> H2.
    for (int j = 0; j < 2; ++j) {
        int n0 = (wave * 2 + j) * 16;
        v8f ca = {}, cb = {};
        gemm_pair(H1, 256, Wt2, 256, 0, 16, n0, 256, lane, ca, cb);
        store_tile(H2, 256, 0, n0, lane, ca, b2, true);
        store_tile(H2, 256, 16, n0, lane, cb, b2, true);
    }
    __syncthreads();

    // GEMM3: (32x256)x(256x64) then sum rows; sum of 32 biases folded in.
    if (wave < 4) {
        int n0 = wave * 16;
        v8f ca = {}, cb = {};
        gemm_pair(H2, 256, Wt3, 256, 0, 16, n0, 256, lane, ca, cb);
        float s = 0.f;
#pragma unroll
        for (int r = 0; r < 8; ++r) s += ca[r] + cb[r];
        s += __shfl_xor(s, 16, 32);      // combine the two row-halves
        if (lane < 16) {
            int n = n0 + lane;
            feH[(size_t)g * 64 + n] = (half_t)(s + (float)NDATA_ * b3[n]);
        }
    }
}

// ---------------------------------------------------------------------------
// Phase 2: per-edge MLP. 64 edges per block.
// ---------------------------------------------------------------------------
__global__ __launch_bounds__(256)
void edge_mlp(const float* __restrict__ edge_attr,
              const long long* __restrict__ edge_index,
              const long long* __restrict__ batch,
              const half_t* __restrict__ feH,
              const half_t* __restrict__ Wt1, const half_t* __restrict__ Wt2,
              const half_t* __restrict__ Wt3,
              const float* __restrict__ b1, const float* __restrict__ b2,
              const float* __restrict__ b3,
              float* __restrict__ out) {
    extern __shared__ char smem[];
    half_t* Ah = (half_t*)smem;          // 64 x 128
    half_t* H1 = Ah + 64 * 128;          // 64 x 256
    half_t* H2 = H1 + 64 * 256;          // 64 x 256
    int* gsrcS = (int*)(H2 + 64 * 256);  // 64
    int* sameS = gsrcS + 64;             // 64

    int tid = threadIdx.x, lane = tid & 31, wave = tid >> 5;
    long long e0 = (long long)blockIdx.x * 64;

    if (tid < 64) {
        long long e = e0 + tid;
        long long src = edge_index[e];
        long long dst = edge_index[(long long)E_ + e];
        long long gs = batch[src];
        long long gd = batch[dst];
        gsrcS[tid] = (int)gs;
        sameS[tid] = (gs == gd) ? 1 : 0;
    }
    __syncthreads();

    // A = [fe(g_src) | edge_attr]  (64 x 128, fp16)
    for (int idx = tid; idx < 64 * 128; idx += 256) {
        int i = idx >> 7, c = idx & 127;
        half_t v;
        if (c < 64) v = feH[(size_t)gsrcS[i] * 64 + c];
        else        v = (half_t)edge_attr[(e0 + i) * 64 + (c - 64)];
        Ah[i * 128 + c] = v;
    }
    __syncthreads();

    // GEMM1: 64x128 -> 64x256. Each wave: 2 col tiles, all 4 row tiles each.
    for (int j = 0; j < 2; ++j) {
        int n0 = (wave * 2 + j) * 16;
        v8f c0 = {}, c1 = {}, c2 = {}, c3 = {};
        gemm_quad(Ah, 128, Wt1, 128, n0, 128, lane, c0, c1, c2, c3);
        store_tile(H1, 256, 0,  n0, lane, c0, b1, true);
        store_tile(H1, 256, 16, n0, lane, c1, b1, true);
        store_tile(H1, 256, 32, n0, lane, c2, b1, true);
        store_tile(H1, 256, 48, n0, lane, c3, b1, true);
    }
    __syncthreads();

    // GEMM2: 64x256 -> 64x256.
    for (int j = 0; j < 2; ++j) {
        int n0 = (wave * 2 + j) * 16;
        v8f c0 = {}, c1 = {}, c2 = {}, c3 = {};
        gemm_quad(H1, 256, Wt2, 256, n0, 256, lane, c0, c1, c2, c3);
        store_tile(H2, 256, 0,  n0, lane, c0, b2, true);
        store_tile(H2, 256, 16, n0, lane, c1, b2, true);
        store_tile(H2, 256, 32, n0, lane, c2, b2, true);
        store_tile(H2, 256, 48, n0, lane, c3, b2, true);
    }
    __syncthreads();

    // GEMM3: 64x256 -> 64x64; masked store. 4 colT x 2 rowPairs = 8 waves.
    {
        int n0 = (wave >> 1) * 16, m0 = (wave & 1) * 32;
        v8f ca = {}, cb = {};
        gemm_pair(H2, 256, Wt3, 256, m0, m0 + 16, n0, 256, lane, ca, cb);
        int hl = lane >> 4, mr = lane & 15;
        int n = n0 + mr;
        float bv = b3[n];
#pragma unroll
        for (int r = 0; r < 8; ++r) {
            int i = m0 + r + hl * 8;
            long long e = e0 + i;
            float v = ca[r] + bv;
            if (!sameS[i]) v = edge_attr[e * 64 + n];
            out[e * 64 + n] = v;
        }
#pragma unroll
        for (int r = 0; r < 8; ++r) {
            int i = m0 + 16 + r + hl * 8;
            long long e = e0 + i;
            float v = cb[r] + bv;
            if (!sameS[i]) v = edge_attr[e * 64 + n];
            out[e * 64 + n] = v;
        }
    }
}

// ---------------------------------------------------------------------------
extern "C" void kernel_launch(void* const* d_in, const int* in_sizes, int n_in,
                              void* d_out, int out_size, void* d_ws, size_t ws_size,
                              hipStream_t stream) {
    const float* x         = (const float*)d_in[0];
    const float* edge_attr = (const float*)d_in[1];
    const float* nw1 = (const float*)d_in[2];
    const float* nb1 = (const float*)d_in[3];
    const float* nw2 = (const float*)d_in[4];
    const float* nb2 = (const float*)d_in[5];
    const float* nw3 = (const float*)d_in[6];
    const float* nb3 = (const float*)d_in[7];
    const float* ew1 = (const float*)d_in[8];
    const float* eb1 = (const float*)d_in[9];
    const float* ew2 = (const float*)d_in[10];
    const float* eb2 = (const float*)d_in[11];
    const float* ew3 = (const float*)d_in[12];
    const float* eb3 = (const float*)d_in[13];
    const long long* edge_index = (const long long*)d_in[14];
    const long long* batch      = (const long long*)d_in[15];

    half_t* ws   = (half_t*)d_ws;
    half_t* nWt1 = ws;                   // 256*128
    half_t* nWt2 = nWt1 + 256 * 128;     // 256*256
    half_t* nWt3 = nWt2 + 256 * 256;     // 64*256
    half_t* eWt1 = nWt3 + 64 * 256;      // 256*128
    half_t* eWt2 = eWt1 + 256 * 128;     // 256*256
    half_t* eWt3 = eWt2 + 256 * 256;     // 64*256
    half_t* feH  = eWt3 + 64 * 256;      // G*64

    prep_wt<<<(128 * 256 + 255) / 256, 256, 0, stream>>>(nw1, nWt1, 128, 256);
    prep_wt<<<(256 * 256 + 255) / 256, 256, 0, stream>>>(nw2, nWt2, 256, 256);
    prep_wt<<<(256 * 64 + 255) / 256, 256, 0, stream>>>(nw3, nWt3, 256, 64);
    prep_wt<<<(128 * 256 + 255) / 256, 256, 0, stream>>>(ew1, eWt1, 128, 256);
    prep_wt<<<(256 * 256 + 255) / 256, 256, 0, stream>>>(ew2, eWt2, 256, 256);
    prep_wt<<<(256 * 64 + 255) / 256, 256, 0, stream>>>(ew3, eWt3, 256, 64);

    size_t lds1 = (size_t)(32 * 128 + 32 * 256 + 32 * 256) * sizeof(half_t);
    node_mlp<<<G_, 256, lds1, stream>>>(x, nWt1, nWt2, nWt3, nb1, nb2, nb3, feH);

    size_t lds2 = (size_t)(64 * 128 + 64 * 256 + 64 * 256) * sizeof(half_t)
                + 128 * sizeof(int);
    edge_mlp<<<E_ / 64, 256, lds2, stream>>>(edge_attr, edge_index, batch, feH,
                                             eWt1, eWt2, eWt3, eb1, eb2, eb3,
                                             (float*)d_out);
}
